// ContextualCouplingTSA_46385646797312
// MI455X (gfx1250) — matrix operand order repaired and checked by reading, not compiled, not run on hardware
//
#include <hip/hip_runtime.h>
#include <math.h>

#define DMODEL 1024
#define DCH    512
#define DHEAD  64
#define NHEADS 8
#define BATCH  2
#define SEQ    2048
#define BL     (BATCH*SEQ)   // 4096

typedef _Float16 half_t;
typedef _Float16 v16h  __attribute__((ext_vector_type(16)));
typedef _Float16 half8 __attribute__((ext_vector_type(8)));
typedef float    v8f   __attribute__((ext_vector_type(8)));

union FragH { v16h v; half8 h[2]; };

// A-matrix fragment (16x32, row-major source, ld in elements).
// Lanes 0-15: row = lane, elements = K{0..7, 16..23}; lanes 16-31: K{8..15, 24..31}.
__device__ __forceinline__ v16h load_a(const half_t* A, int lda, int m0, int k0, int lane) {
    FragH f;
    const int r = m0 + (lane & 15);
    const int k = k0 + ((lane & 16) ? 8 : 0);
    f.h[0] = *(const half8*)(A + (size_t)r * lda + k);
    f.h[1] = *(const half8*)(A + (size_t)r * lda + k + 16);
    return f.v;
}

// B-matrix fragment from B-transposed (row-major [N][K]) source.
// Lane = column n (mod 16); lanes 0-15 hold K{0..7, 16..23}; lanes 16-31 hold K{8..15, 24..31}.
__device__ __forceinline__ v16h load_bt(const half_t* Bt, int ldb, int n0, int k0, int lane) {
    FragH f;
    const int c = n0 + (lane & 15);
    const int k = k0 + ((lane & 16) ? 8 : 0);
    f.h[0] = *(const half8*)(Bt + (size_t)c * ldb + k);
    f.h[1] = *(const half8*)(Bt + (size_t)c * ldb + k + 16);
    return f.v;
}

#define WMMA_F16(a, b, c) \
    __builtin_amdgcn_wmma_f32_16x16x32_f16(false, (a), false, (b), (short)0, (c), false, false)

// ---------------- conversion kernels ----------------
__global__ void k_cvt(const float* __restrict__ s, half_t* __restrict__ d, int n) {
    int i = blockIdx.x * blockDim.x + threadIdx.x;
    if (i < n) d[i] = (half_t)s[i];
}

__global__ void k_cvt_x1(const float* __restrict__ src, half_t* __restrict__ d) {
    int i = blockIdx.x * blockDim.x + threadIdx.x;   // BL*DCH elements
    if (i >= BL * DCH) return;
    int r = i >> 9, c = i & 511;
    d[i] = (half_t)src[(size_t)r * DMODEL + c];
}

// ---------------- saliency conv1d (kernel 3, SAME) ----------------
__global__ void k_sal(const float* __restrict__ src, const float* __restrict__ cw,
                      const float* __restrict__ cb, float* __restrict__ sal) {
    int i = blockIdx.x * blockDim.x + threadIdx.x;   // BL*NHEADS
    if (i >= BL * NHEADS) return;
    int h = i & 7, bl = i >> 3;
    int b = bl >> 11, l = bl & 2047;
    const float* x = src + (size_t)bl * DMODEL;      // x1 row (first 512 cols)
    float acc = cb[h];
    for (int c = 0; c < DCH; ++c) {
        const float* wp = cw + ((size_t)h * DCH + c) * 3;
        float xm = (l > 0)       ? src[((size_t)bl - 1) * DMODEL + c] : 0.f;
        float xc = x[c];
        float xp = (l < SEQ - 1) ? src[((size_t)bl + 1) * DMODEL + c] : 0.f;
        acc += wp[0] * xm + wp[1] * xc + wp[2] * xp;
    }
    sal[((size_t)b * NHEADS + h) * SEQ + l] = acc;
}

// ---------------- Q/K projection GEMM: one wave = 16x64 tile (A reused 4x) ----------------
__global__ void k_gemm_qk(const half_t* __restrict__ X1h, const half_t* __restrict__ Wh,
                          const float* __restrict__ bias, half_t* __restrict__ outH,
                          half_t* __restrict__ KT) {
    int lane = threadIdx.x;
    int nt = blockIdx.x;          // 64-col tile: 0..7
    int mt = blockIdx.y;          // 16-row tile: 0..255
    v8f acc0 = {}, acc1 = {}, acc2 = {}, acc3 = {};
    for (int kc = 0; kc < DCH; kc += 32) {
        v16h a  = load_a(X1h, DCH, mt * 16, kc, lane);
        v16h b0 = load_bt(Wh, DCH, nt * 64 + 0,  kc, lane);
        v16h b1 = load_bt(Wh, DCH, nt * 64 + 16, kc, lane);
        v16h b2 = load_bt(Wh, DCH, nt * 64 + 32, kc, lane);
        v16h b3 = load_bt(Wh, DCH, nt * 64 + 48, kc, lane);
        acc0 = WMMA_F16(a, b0, acc0);
        acc1 = WMMA_F16(a, b1, acc1);
        acc2 = WMMA_F16(a, b2, acc2);
        acc3 = WMMA_F16(a, b3, acc3);
    }
    v8f accs[4] = {acc0, acc1, acc2, acc3};
#pragma unroll
    for (int j = 0; j < 4; ++j) {
        int c = nt * 64 + j * 16 + (lane & 15);
        float bv = bias[c];
        for (int i = 0; i < 8; ++i) {
            int r = mt * 16 + i + ((lane & 16) ? 8 : 0);
            float v = accs[j][i] + bv;
            outH[(size_t)r * DCH + c] = (half_t)v;
            if (KT) {  // scatter transposed copy: KT[b][h][d][l]
                int b_ = r >> 11, l_ = r & 2047, h_ = c >> 6, d_ = c & 63;
                KT[(((size_t)(b_ * NHEADS + h_)) * DHEAD + d_) * SEQ + l_] = (half_t)v;
            }
        }
    }
}

// ---------------- scores + online row max/sum ----------------
__global__ void k_scores(const half_t* __restrict__ Qh, const half_t* __restrict__ Kh,
                         const float* __restrict__ sal, float* __restrict__ attn,
                         float* __restrict__ rowm, float* __restrict__ rows) {
    int lane = threadIdx.x;
    int qt = blockIdx.x;          // 0..127
    int bh = blockIdx.y;          // 0..15
    int b = bh >> 3, h = bh & 7;
    const half_t* Aq = Qh + (size_t)b * SEQ * DCH + h * DHEAD;
    const half_t* Bk = Kh + (size_t)b * SEQ * DCH + h * DHEAD;
    v16h a0 = load_a(Aq, DCH, qt * 16, 0, lane);
    v16h a1 = load_a(Aq, DCH, qt * 16, 32, lane);
    const float* salh = sal + (size_t)bh * SEQ;
    float* arow = attn + (size_t)bh * SEQ * SEQ;
    float m[8], s[8];
    for (int i = 0; i < 8; ++i) { m[i] = -3.0e38f; s[i] = 0.f; }
    for (int nt = 0; nt < SEQ / 16; ++nt) {
        v16h b0 = load_bt(Bk, DCH, nt * 16, 0, lane);
        v16h b1 = load_bt(Bk, DCH, nt * 16, 32, lane);
        v8f acc = {};
        acc = WMMA_F16(a0, b0, acc);
        acc = WMMA_F16(a1, b1, acc);
        int n = nt * 16 + (lane & 15);
        float sv = salh[n];
        for (int i = 0; i < 8; ++i) {
            int r = qt * 16 + i + ((lane & 16) ? 8 : 0);
            float x = acc[i] * 0.125f + sv;               // 1/sqrt(64)
            arow[(size_t)r * SEQ + n] = x;                // raw score
            float nm = fmaxf(m[i], x);
            s[i] = s[i] * __expf(m[i] - nm) + __expf(x - nm);
            m[i] = nm;
        }
    }
    // combine across the 16 lanes that share each row
    for (int i = 0; i < 8; ++i) {
        for (int mask = 1; mask <= 8; mask <<= 1) {
            float om = __shfl_xor(m[i], mask, 32);
            float os = __shfl_xor(s[i], mask, 32);
            float nm = fmaxf(m[i], om);
            s[i] = s[i] * __expf(m[i] - nm) + os * __expf(om - nm);
            m[i] = nm;
        }
        if ((lane & 15) == 0) {
            int r = qt * 16 + i + ((lane & 16) ? 8 : 0);
            rowm[(size_t)bh * SEQ + r] = m[i];
            rows[(size_t)bh * SEQ + r] = s[i];
        }
    }
}

// ---------------- normalize attn in-place + context = attn @ K ----------------
__global__ void k_normctx(const half_t* __restrict__ KT, float* __restrict__ attn,
                          const float* __restrict__ rowm, const float* __restrict__ rows,
                          half_t* __restrict__ Ctx) {
    __shared__ __align__(16) half_t sA[16 * 32];
    __shared__ float sM[16], sS[16];
    int lane = threadIdx.x;
    int qt = blockIdx.x, bh = blockIdx.y;
    int b = bh >> 3, h = bh & 7;
    float* arow = attn + (size_t)bh * SEQ * SEQ + (size_t)qt * 16 * SEQ;
    if (lane < 16) {
        sM[lane] = rowm[(size_t)bh * SEQ + qt * 16 + lane];
        sS[lane] = 1.0f / rows[(size_t)bh * SEQ + qt * 16 + lane];
    }
    __syncthreads();
    const half_t* Bt = KT + (size_t)bh * DHEAD * SEQ;   // [64][2048] (feature-major)
    v8f acc0 = {}, acc1 = {}, acc2 = {}, acc3 = {};
    for (int nc = 0; nc < SEQ; nc += 32) {
        for (int e = lane; e < 16 * 32; e += 32) {
            int r = e >> 5, kk = e & 31;
            size_t idx = (size_t)r * SEQ + nc + kk;
            float x = arow[idx];
            float v = __expf(x - sM[r]) * sS[r];
            arow[idx] = v;                 // final normalized attn (f32 output)
            sA[r * 32 + kk] = (half_t)v;
        }
        __syncthreads();
        v16h a  = load_a(sA, 32, 0, 0, lane);
        v16h b0 = load_bt(Bt, SEQ, 0,  nc, lane);
        v16h b1 = load_bt(Bt, SEQ, 16, nc, lane);
        v16h b2 = load_bt(Bt, SEQ, 32, nc, lane);
        v16h b3 = load_bt(Bt, SEQ, 48, nc, lane);
        acc0 = WMMA_F16(a, b0, acc0);
        acc1 = WMMA_F16(a, b1, acc1);
        acc2 = WMMA_F16(a, b2, acc2);
        acc3 = WMMA_F16(a, b3, acc3);
        __syncthreads();
    }
    int cbase = h * DHEAD;
    for (int i = 0; i < 8; ++i) {
        int r = qt * 16 + i + ((lane & 16) ? 8 : 0);
        int c = lane & 15;
        size_t row = (size_t)(b * SEQ + r) * DCH;
        Ctx[row + cbase + 0  + c] = (half_t)acc0[i];
        Ctx[row + cbase + 16 + c] = (half_t)acc1[i];
        Ctx[row + cbase + 32 + c] = (half_t)acc2[i];
        Ctx[row + cbase + 48 + c] = (half_t)acc3[i];
    }
}

// ---------------- gate GEMM + y2 mix: one wave = 16x64 tile ----------------
__global__ void k_gate(const half_t* __restrict__ Ctx, const half_t* __restrict__ Wsth,
                       const float* __restrict__ bst, const float* __restrict__ src,
                       half_t* __restrict__ Y2) {
    int lane = threadIdx.x;
    int nt = blockIdx.x;          // 0..7
    int mt = blockIdx.y;          // 0..255
    v8f acc0 = {}, acc1 = {}, acc2 = {}, acc3 = {};
    for (int kc = 0; kc < DCH; kc += 32) {
        v16h a  = load_a(Ctx, DCH, mt * 16, kc, lane);
        v16h b0 = load_bt(Wsth, DCH, nt * 64 + 0,  kc, lane);
        v16h b1 = load_bt(Wsth, DCH, nt * 64 + 16, kc, lane);
        v16h b2 = load_bt(Wsth, DCH, nt * 64 + 32, kc, lane);
        v16h b3 = load_bt(Wsth, DCH, nt * 64 + 48, kc, lane);
        acc0 = WMMA_F16(a, b0, acc0);
        acc1 = WMMA_F16(a, b1, acc1);
        acc2 = WMMA_F16(a, b2, acc2);
        acc3 = WMMA_F16(a, b3, acc3);
    }
    v8f accs[4] = {acc0, acc1, acc2, acc3};
#pragma unroll
    for (int j = 0; j < 4; ++j) {
        int c = nt * 64 + j * 16 + (lane & 15);
        float bv = bst[c];
        for (int i = 0; i < 8; ++i) {
            int r = mt * 16 + i + ((lane & 16) ? 8 : 0);
            float g = 1.0f / (1.0f + __expf(-(accs[j][i] + bv)));
            float x2 = src[(size_t)r * DMODEL + DCH + c];
            float y = g * tanhf(x2) + (1.0f - g) * x2;
            Y2[(size_t)r * DCH + c] = (half_t)y;
        }
    }
}

// ---------------- output GEMM: concat([x1,y2]) @ Wout^T + bout, 16x64 per wave ----------------
__global__ void k_out(const half_t* __restrict__ X1h, const half_t* __restrict__ Y2h,
                      const half_t* __restrict__ Wouth, const float* __restrict__ bout,
                      float* __restrict__ out) {
    int lane = threadIdx.x;
    int nt = blockIdx.x;          // 0..15
    int mt = blockIdx.y;          // 0..255
    v8f acc0 = {}, acc1 = {}, acc2 = {}, acc3 = {};
    for (int kc = 0; kc < DMODEL; kc += 32) {
        const half_t* A = (kc < DCH) ? X1h : Y2h;
        int kk = (kc < DCH) ? kc : kc - DCH;
        v16h a  = load_a(A, DCH, mt * 16, kk, lane);
        v16h b0 = load_bt(Wouth, DMODEL, nt * 64 + 0,  kc, lane);
        v16h b1 = load_bt(Wouth, DMODEL, nt * 64 + 16, kc, lane);
        v16h b2 = load_bt(Wouth, DMODEL, nt * 64 + 32, kc, lane);
        v16h b3 = load_bt(Wouth, DMODEL, nt * 64 + 48, kc, lane);
        acc0 = WMMA_F16(a, b0, acc0);
        acc1 = WMMA_F16(a, b1, acc1);
        acc2 = WMMA_F16(a, b2, acc2);
        acc3 = WMMA_F16(a, b3, acc3);
    }
    v8f accs[4] = {acc0, acc1, acc2, acc3};
#pragma unroll
    for (int j = 0; j < 4; ++j) {
        int c = nt * 64 + j * 16 + (lane & 15);
        float bv = bout[c];
        for (int i = 0; i < 8; ++i) {
            int r = mt * 16 + i + ((lane & 16) ? 8 : 0);
            out[(size_t)r * DMODEL + c] = accs[j][i] + bv;
        }
    }
}

extern "C" void kernel_launch(void* const* d_in, const int* in_sizes, int n_in,
                              void* d_out, int out_size, void* d_ws, size_t ws_size,
                              hipStream_t stream) {
    const float* src  = (const float*)d_in[0];
    const float* Wq   = (const float*)d_in[1];
    const float* bq   = (const float*)d_in[2];
    const float* Wk   = (const float*)d_in[3];
    const float* bk   = (const float*)d_in[4];
    const float* cw   = (const float*)d_in[5];
    const float* cb   = (const float*)d_in[6];
    const float* Wst  = (const float*)d_in[7];
    const float* bst  = (const float*)d_in[8];
    const float* Wout = (const float*)d_in[9];
    const float* bout = (const float*)d_in[10];

    float* out  = (float*)d_out;
    float* attn = out + (size_t)BL * DMODEL;   // second output, 2*8*2048*2048 f32

    char* w = (char*)d_ws;
    auto take = [&](size_t bytes) { void* p = (void*)w; w += bytes; return p; };
    half_t* X1h   = (half_t*)take((size_t)BL * DCH * 2);
    half_t* Wqh   = (half_t*)take((size_t)DCH * DCH * 2);
    half_t* Wkh   = (half_t*)take((size_t)DCH * DCH * 2);
    half_t* Wsth  = (half_t*)take((size_t)DCH * DCH * 2);
    half_t* Wouth = (half_t*)take((size_t)DMODEL * DMODEL * 2);
    half_t* Qh    = (half_t*)take((size_t)BL * DCH * 2);
    half_t* Kh    = (half_t*)take((size_t)BL * DCH * 2);
    half_t* KTh   = (half_t*)take((size_t)BATCH * NHEADS * DHEAD * SEQ * 2);
    half_t* Ctxh  = (half_t*)take((size_t)BL * DCH * 2);
    half_t* Y2h   = (half_t*)take((size_t)BL * DCH * 2);
    float*  sal   = (float*)take((size_t)BATCH * NHEADS * SEQ * 4);
    float*  rowm  = (float*)take((size_t)BATCH * NHEADS * SEQ * 4);
    float*  rows  = (float*)take((size_t)BATCH * NHEADS * SEQ * 4);

    // 1) convert activations/weights to f16
    k_cvt_x1<<<(BL * DCH + 255) / 256, 256, 0, stream>>>(src, X1h);
    k_cvt<<<(DCH * DCH + 255) / 256, 256, 0, stream>>>(Wq, Wqh, DCH * DCH);
    k_cvt<<<(DCH * DCH + 255) / 256, 256, 0, stream>>>(Wk, Wkh, DCH * DCH);
    k_cvt<<<(DCH * DCH + 255) / 256, 256, 0, stream>>>(Wst, Wsth, DCH * DCH);
    k_cvt<<<(DMODEL * DMODEL + 255) / 256, 256, 0, stream>>>(Wout, Wouth, DMODEL * DMODEL);

    // 2) saliency conv
    k_sal<<<(BL * NHEADS + 255) / 256, 256, 0, stream>>>(src, cw, cb, sal);

    // 3) Q and K projections (K also stores a transposed per-head copy)
    k_gemm_qk<<<dim3(DCH / 64, BL / 16), 32, 0, stream>>>(X1h, Wqh, bq, Qh, (half_t*)nullptr);
    k_gemm_qk<<<dim3(DCH / 64, BL / 16), 32, 0, stream>>>(X1h, Wkh, bk, Kh, KTh);

    // 4) raw scores + online softmax stats
    k_scores<<<dim3(SEQ / 16, BATCH * NHEADS), 32, 0, stream>>>(Qh, Kh, sal, attn, rowm, rows);

    // 5) normalize attn in place + context = attn @ K
    k_normctx<<<dim3(SEQ / 16, BATCH * NHEADS), 32, 0, stream>>>(KTh, attn, rowm, rows, Ctxh);

    // 6) gate + y2 mix
    k_gate<<<dim3(DCH / 64, BL / 16), 32, 0, stream>>>(Ctxh, Wsth, bst, src, Y2h);

    // 7) output GEMM
    k_out<<<dim3(DMODEL / 64, BL / 16), 32, 0, stream>>>(X1h, Y2h, Wouth, bout, out);
}
